// ChildSumTreeLSTM_65077344469021
// MI455X (gfx1250) — matrix operand order repaired and checked by reading, not compiled, not run on hardware
//
#include <hip/hip_runtime.h>
#include <hip/hip_bf16.h>
#include <math.h>

// ChildSum TreeLSTM, N=4096 nodes, H=1024.
// Level-scheduled tree (expected depth ~22, bound 64): the sequential scan
// becomes per-level batched GEMMs on v_wmma_f32_16x16x32_bf16 using split
// precision (bf16 hi/lo, 3 WMMAs ~ fp32). All operands are PRE-SPLIT into
// packed bf16 rows [hi(0..H-1) | lo(0..H-1)] so the GEMM inner loop is pure
// loads + WMMA (no per-tile cvt). Working set (~144MB) is L2-resident (192MB).

#define N_NODES 4096
#define HDIM    1024
#define LMAX    64

typedef __attribute__((ext_vector_type(16))) __bf16 v16bf;
typedef __attribute__((ext_vector_type(8)))  __bf16 v8bf;
typedef __attribute__((ext_vector_type(8)))  float  v8f;

static __device__ __forceinline__ float sigf(float x) {
    return 1.0f / (1.0f + __expf(-x));
}

// Fragment layouts used here (contiguous k per lane half, same for A and B; wave32):
//   A(16x32 bf16): lane<16 -> M=lane,    K = k0+{0..15}
//                  lane>=16 -> M=lane-16, K = k0+{16..31}
//   B(32x16):      N=lane&15, K = k0+{0..15} (lanes<16) / {16..31} (lanes>=16)
//   C(16x16 f32):  VGPR r -> M=r (lanes<16) / r+8 (lanes>=16), N=lane&15
// Packed rows are [row][2H]: hi half then lo half, both contiguous in K, so a
// lane's B fragment is one 32B load and an A fragment is two 16B loads.

static __device__ __forceinline__ v16bf load_a_frag(const __bf16* __restrict__ row,
                                                    int k0, int aoff) {
    union { v16bf v; v8bf h[2]; } u;
    u.h[0] = *(const v8bf*)(row + k0 + aoff);
    u.h[1] = *(const v8bf*)(row + k0 + 8 + aoff);
    return u.v;
}

static __device__ __forceinline__ v16bf load_b_frag(const __bf16* __restrict__ row,
                                                    int k0, int boff) {
    return *(const v16bf*)(row + k0 + boff);
}

// ---------------------------------------------------------------------------
// Phase A: Wx_g = x @ W_g.T + b_g  (gate = blockIdx.z). 16x64 strip per wave.
// ---------------------------------------------------------------------------
__global__ __launch_bounds__(128)
void wx_gemm(const __bf16* __restrict__ xs,   // packed x   [N][2H]
             const __bf16* __restrict__ Ws,   // packed W_i..W_f, 4 x [H][2H]
             const float* __restrict__ b0, const float* __restrict__ b1,
             const float* __restrict__ b2, const float* __restrict__ b3,
             float* __restrict__ O0, float* __restrict__ O1,
             float* __restrict__ O2, float* __restrict__ O3) {
    const int g = blockIdx.z;
    const float* bias = (g == 0) ? b0 : (g == 1) ? b1 : (g == 2) ? b2 : b3;
    float*       O    = (g == 0) ? O0 : (g == 1) ? O1 : (g == 2) ? O2 : O3;
    const __bf16* B = Ws + (size_t)g * 2 * HDIM * HDIM;

    const int lane = threadIdx.x & 31;
    const int wave = threadIdx.x >> 5;
    const int nbase = (blockIdx.x * 4 + wave) * 64;   // 16x64 strip
    const int m0 = blockIdx.y * 16;
    const int hl   = (lane < 16) ? 0 : 1;
    const int l15  = lane & 15;
    const int aoff = hl ? 16 : 0;
    const int boff = hl ? 16 : 0;

    const __bf16* ar  = xs + (size_t)(m0 + l15) * 2 * HDIM;
    const __bf16* br0 = B  + (size_t)(nbase + l15) * 2 * HDIM;

    v8f acc[4] = {{}, {}, {}, {}};
    for (int k0 = 0; k0 < HDIM; k0 += 32) {
        v16bf ah = load_a_frag(ar, k0, aoff);
        v16bf al = load_a_frag(ar + HDIM, k0, aoff);
        __builtin_prefetch(ar + k0 + 64, 0, 3);   // global_prefetch_b8
#pragma unroll
        for (int nt = 0; nt < 4; nt++) {
            const __bf16* br = br0 + (size_t)(nt * 16) * 2 * HDIM;
            v16bf bh = load_b_frag(br, k0, boff);
            v16bf bl = load_b_frag(br + HDIM, k0, boff);
            acc[nt] = __builtin_amdgcn_wmma_f32_16x16x32_bf16(false, ah, false, bh, (short)0, acc[nt], false, false);
            acc[nt] = __builtin_amdgcn_wmma_f32_16x16x32_bf16(false, ah, false, bl, (short)0, acc[nt], false, false);
            acc[nt] = __builtin_amdgcn_wmma_f32_16x16x32_bf16(false, al, false, bh, (short)0, acc[nt], false, false);
        }
    }
    const int rbase = m0 + (hl ? 8 : 0);
#pragma unroll
    for (int nt = 0; nt < 4; nt++) {
        const int n = nbase + nt * 16 + l15;
        const float bv = bias[n];
#pragma unroll
        for (int r = 0; r < 8; r++)
            O[(size_t)(rbase + r) * HDIM + n] = acc[nt][r] + bv;
    }
}

// ---------------------------------------------------------------------------
// Per-level batched GEMM with row gather: Out[order[t]] (+)= A[order[t]] @ U.T
// A: packed bf16 rows, stride 2H. B: Us + (ugate0+z)*[H][2H].
// Rows past the level end contribute zero A fragments and skip stores.
// ---------------------------------------------------------------------------
__global__ __launch_bounds__(128)
void level_gemm(const __bf16* __restrict__ A,
                const __bf16* __restrict__ Us, int ugate0,
                float* __restrict__ O0, float* __restrict__ O1,
                float* __restrict__ O2,
                int accumulate,
                const int* __restrict__ order,
                const int* __restrict__ levelStart, int lvl) {
    const int ls = levelStart[lvl];
    const int le = levelStart[lvl + 1];
    const int rows = le - ls;
    if ((int)blockIdx.y * 16 >= rows) return;

    const int g = blockIdx.z;
    float* O = (g == 0) ? O0 : (g == 1) ? O1 : O2;
    const __bf16* B = Us + (size_t)(ugate0 + g) * 2 * HDIM * HDIM;

    const int lane = threadIdx.x & 31;
    const int wave = threadIdx.x >> 5;
    const int nbase = (blockIdx.x * 4 + wave) * 64;
    const int hl   = (lane < 16) ? 0 : 1;
    const int l15  = lane & 15;
    const int aoff = hl ? 16 : 0;
    const int boff = hl ? 16 : 0;

    const int t = ls + blockIdx.y * 16 + l15;
    const bool avalid = (t < le);
    const int j = avalid ? order[t] : 0;
    const __bf16* ar  = A + (size_t)j * 2 * HDIM;
    const __bf16* br0 = B + (size_t)(nbase + l15) * 2 * HDIM;

    v8f acc[4] = {{}, {}, {}, {}};
    for (int k0 = 0; k0 < HDIM; k0 += 32) {
        v16bf ah = {}, al = {};
        if (avalid) {
            ah = load_a_frag(ar, k0, aoff);
            al = load_a_frag(ar + HDIM, k0, aoff);
        }
#pragma unroll
        for (int nt = 0; nt < 4; nt++) {
            const __bf16* br = br0 + (size_t)(nt * 16) * 2 * HDIM;
            v16bf bh = load_b_frag(br, k0, boff);
            v16bf bl = load_b_frag(br + HDIM, k0, boff);
            acc[nt] = __builtin_amdgcn_wmma_f32_16x16x32_bf16(false, ah, false, bh, (short)0, acc[nt], false, false);
            acc[nt] = __builtin_amdgcn_wmma_f32_16x16x32_bf16(false, ah, false, bl, (short)0, acc[nt], false, false);
            acc[nt] = __builtin_amdgcn_wmma_f32_16x16x32_bf16(false, al, false, bh, (short)0, acc[nt], false, false);
        }
    }
    const int tb = ls + blockIdx.y * 16 + (hl ? 8 : 0);
#pragma unroll
    for (int r = 0; r < 8; r++) {
        int t2 = tb + r;
        if (t2 < le) {
            size_t row = (size_t)order[t2] * HDIM;
#pragma unroll
            for (int nt = 0; nt < 4; nt++) {
                size_t idx = row + nbase + nt * 16 + l15;
                if (accumulate) O[idx] += acc[nt][r];
                else            O[idx]  = acc[nt][r];
            }
        }
    }
}

// Split a f32 matrix into packed bf16 hi/lo rows: dst[r][0..H)=hi, [H..2H)=lo.
__global__ void split_matrix(const float* __restrict__ src, __bf16* __restrict__ dst,
                             int rows) {
    size_t total = (size_t)rows * HDIM;
    size_t i = (size_t)blockIdx.x * blockDim.x + threadIdx.x;
    size_t stride = (size_t)gridDim.x * blockDim.x;
    for (; i < total; i += stride) {
        size_t r = i >> 10, k = i & (HDIM - 1);
        float v = src[i];
        __bf16 h = (__bf16)v;
        dst[r * 2 * HDIM + k] = h;
        dst[r * 2 * HDIM + HDIM + k] = (__bf16)(v - (float)h);
    }
}

// Split the (now final) hsum rows of one level into packed bf16 rows.
__global__ void split_level_rows(const float* __restrict__ src, __bf16* __restrict__ dst,
                                 const int* __restrict__ order,
                                 const int* __restrict__ levelStart, int lvl) {
    int ls = levelStart[lvl], le = levelStart[lvl + 1];
    int t = ls + blockIdx.x;
    if (t >= le) return;
    int j = order[t];
    const float* s = src + (size_t)j * HDIM;
    __bf16* d = dst + (size_t)j * 2 * HDIM;
    for (int n = threadIdx.x; n < HDIM; n += blockDim.x) {
        float v = s[n];
        __bf16 h = (__bf16)v;
        d[n] = h;
        d[HDIM + n] = (__bf16)(v - (float)h);
    }
}

// Pointwise gates for one level: c = sig(i)*tanh(u) + fcsum; h = sig(o)*tanh(c).
// Also packs h into bf16 hi/lo, overwriting the dead Gi(=Wxi) row in place
// (sync separates the f32 reads from the bf16 overwrite).
__global__ __launch_bounds__(256)
void pointwise(const float* __restrict__ Gi, const float* __restrict__ Go,
               const float* __restrict__ Gu, const float* __restrict__ fcsum,
               const int* __restrict__ order,
               const int* __restrict__ levelStart, int lvl,
               float* __restrict__ h, float* __restrict__ c,
               __bf16* __restrict__ hsplit /* == (__bf16*)Gi */) {
    int ls = levelStart[lvl], le = levelStart[lvl + 1];
    int t = ls + blockIdx.x;
    if (t >= le) return;
    size_t j = (size_t)order[t];
    float hv4[HDIM / 256];
#pragma unroll
    for (int q = 0; q < HDIM / 256; q++) {
        int n = threadIdx.x + q * 256;
        size_t idx = j * HDIM + n;
        float iv = sigf(Gi[idx]);
        float ov = sigf(Go[idx]);
        float uv = tanhf(Gu[idx]);
        float cv = iv * uv + fcsum[idx];
        float hv = ov * tanhf(cv);
        c[idx] = cv;
        h[idx] = hv;
        hv4[q] = hv;
    }
    __syncthreads();   // all Gi reads complete before the row is reused as bf16
    __bf16* d = hsplit + j * 2 * HDIM;
#pragma unroll
    for (int q = 0; q < HDIM / 256; q++) {
        int n = threadIdx.x + q * 256;
        __bf16 hb = (__bf16)hv4[q];
        d[n] = hb;
        d[HDIM + n] = (__bf16)(hv4[q] - (float)hb);
    }
}

// Scatter into parents: hsum[p]+=h; fcsum[p]+=sig(Wxf[p]+F[j])*c
__global__ void scatter(const float* __restrict__ h, const float* __restrict__ c,
                        const float* __restrict__ F, const float* __restrict__ Wxf,
                        const int* __restrict__ head,
                        float* __restrict__ hsum, float* __restrict__ fcsum,
                        const int* __restrict__ order,
                        const int* __restrict__ levelStart, int lvl) {
    int ls = levelStart[lvl], le = levelStart[lvl + 1];
    int t = ls + blockIdx.x;
    if (t >= le) return;
    int j = order[t];
    int p = head[j];   // p == N for the root -> pad rows
    for (int n = threadIdx.x; n < HDIM; n += blockDim.x) {
        size_t ji = (size_t)j * HDIM + n;
        size_t pi = (size_t)p * HDIM + n;
        float hv = h[ji], cv = c[ji];
        atomicAdd(&hsum[pi], hv);
        float f = sigf(Wxf[pi] + F[ji]);
        atomicAdd(&fcsum[pi], f * cv);
    }
}

__global__ void zero_fill(float* __restrict__ p, size_t count) {
    size_t i = (size_t)blockIdx.x * blockDim.x + threadIdx.x;
    size_t stride = (size_t)gridDim.x * blockDim.x;
    for (; i < count; i += stride) p[i] = 0.0f;
}

// Single-thread level bucketing: head[j] > j, so one ascending pass finalizes
// level[j] = 1 + max(level of children). Then counting-sort nodes by level.
__global__ void build_levels(const int* __restrict__ head,
                             int* __restrict__ lvl, int* __restrict__ order,
                             int* __restrict__ levelStart, int* __restrict__ cursor) {
    if (threadIdx.x != 0 || blockIdx.x != 0) return;
    for (int j = 0; j < N_NODES; j++) lvl[j] = 0;
    for (int j = 0; j < N_NODES; j++) {
        int p = head[j];
        if (p < N_NODES) {
            int cnd = lvl[j] + 1;
            if (cnd > lvl[p]) lvl[p] = cnd;
        }
    }
    for (int l = 0; l < LMAX; l++) cursor[l] = 0;
    for (int j = 0; j < N_NODES; j++) {
        int l = lvl[j];
        if (l > LMAX - 1) l = LMAX - 1;   // never expected (random tree depth ~22)
        lvl[j] = l;
        cursor[l]++;
    }
    levelStart[0] = 0;
    for (int l = 0; l < LMAX; l++) levelStart[l + 1] = levelStart[l] + cursor[l];
    for (int l = 0; l < LMAX; l++) cursor[l] = levelStart[l];
    for (int j = 0; j < N_NODES; j++) order[cursor[lvl[j]]++] = j;
}

extern "C" void kernel_launch(void* const* d_in, const int* in_sizes, int n_in,
                              void* d_out, int out_size, void* d_ws, size_t ws_size,
                              hipStream_t stream) {
    const float* x    = (const float*)d_in[0];
    const int*   head = (const int*)  d_in[1];
    const float* Wmat[8] = { (const float*)d_in[2], (const float*)d_in[3],
                             (const float*)d_in[4], (const float*)d_in[5],
                             (const float*)d_in[6], (const float*)d_in[7],
                             (const float*)d_in[8], (const float*)d_in[9] };
    const float* b_i  = (const float*)d_in[10];
    const float* b_o  = (const float*)d_in[11];
    const float* b_u  = (const float*)d_in[12];
    const float* b_f  = (const float*)d_in[13];

    float* h_out = (float*)d_out;
    float* c_out = h_out + (size_t)N_NODES * HDIM;

    // Workspace (~144 MB): Wxi|Wxo|Wxu|Wxf(+pad)|hsum(+pad)|fcsum(+pad)|
    //                      xsplit(bf16, reused for hsum rows)|Wsplit(bf16 x8)|ints
    const size_t NH = (size_t)N_NODES * HDIM;
    float* w     = (float*)d_ws;
    float* Wxi   = w;
    float* Wxo   = Wxi + NH;
    float* Wxu   = Wxo + NH;
    float* Wxf   = Wxu + NH;            // (N+1) rows, pad row for root's parent
    float* hsum  = Wxf + NH + HDIM;     // (N+1) rows
    float* fcsum = hsum + NH + HDIM;    // (N+1) rows
    float* F     = Wxu;                 // alias: Wxu[j] is dead after pointwise(level(j))
    __bf16* xsplit = (__bf16*)(fcsum + NH + HDIM);  // [N][2H]; x, later hsum rows
    __bf16* Wsplit = xsplit + 2 * NH;               // 8 x [H][2H]
    int* lvl        = (int*)(Wsplit + (size_t)16 * HDIM * HDIM);
    int* order      = lvl + N_NODES;
    int* levelStart = order + N_NODES;        // LMAX+1 entries
    int* cursor     = levelStart + (LMAX + 1);

    // Zero Wxf pad row + hsum + fcsum (one contiguous range).
    zero_fill<<<512, 256, 0, stream>>>(Wxf + NH, (size_t)HDIM + 2 * (NH + HDIM));
    build_levels<<<1, 1, 0, stream>>>(head, lvl, order, levelStart, cursor);

    // One-time bf16 hi/lo splits: x and all 8 weight matrices.
    split_matrix<<<1024, 256, 0, stream>>>(x, xsplit, N_NODES);
    for (int m = 0; m < 8; m++)
        split_matrix<<<1024, 256, 0, stream>>>(
            Wmat[m], Wsplit + (size_t)m * 2 * HDIM * HDIM, HDIM);

    // Phase A: four input projections as WMMA GEMMs (16x64 strip per wave).
    wx_gemm<<<dim3(HDIM / 256, N_NODES / 16, 4), 128, 0, stream>>>(
        xsplit, Wsplit, b_i, b_o, b_u, b_f, Wxi, Wxo, Wxu, Wxf);

    // Phase B: process tree level by level (each level = batched GEMMs).
    for (int l = 0; l < LMAX; l++) {
        // hsum rows of this level are final -> pack to bf16 (reuses xsplit).
        split_level_rows<<<N_NODES, 256, 0, stream>>>(
            hsum, xsplit, order, levelStart, l);
        // Recurrent gate GEMMs: Wx_g[j] += hsum[j] @ U_g.T for g in {i,o,u}.
        level_gemm<<<dim3(HDIM / 256, N_NODES / 16, 3), 128, 0, stream>>>(
            xsplit, Wsplit, 4, Wxi, Wxo, Wxu, 1, order, levelStart, l);
        pointwise<<<N_NODES, 256, 0, stream>>>(
            Wxi, Wxo, Wxu, fcsum, order, levelStart, l, h_out, c_out,
            (__bf16*)Wxi);
        // Forget-gate GEMM: F[j] = h[j] @ U_f.T (A = packed h in Wxi rows).
        level_gemm<<<dim3(HDIM / 256, N_NODES / 16, 1), 128, 0, stream>>>(
            (const __bf16*)Wxi, Wsplit, 7, F, F, F, 0, order, levelStart, l);
        scatter<<<N_NODES, 256, 0, stream>>>(
            h_out, c_out, F, Wxf, head, hsum, fcsum, order, levelStart, l);
    }
}
